// StateExpandedBlock_82214263980186
// MI455X (gfx1250) — matrix operand order repaired and checked by reading, not compiled, not run on hardware
//
#include <hip/hip_runtime.h>
#include <hip/hip_bf16.h>
#include <stdint.h>

// Problem dims (match reference)
#define BB     4
#define TT     4096
#define DM     1024          // d_model
#define NH     8
#define HD     128
#define MROWS  (BB * TT)     // 16384 token rows
#define NTOT   (3 * DM)      // concatenated f|v|o projection width = 3072
#define FORGET_LB 0.9f
#define RMS_EPS   1e-6f

// GEMM block tile
#define MT 128               // rows per block (8 waves x 16)
#define NT 64                // cols per block (4 wmma subtiles)
#define KT 32                // K per stage
#define KTP 40               // padded A row stride in elements (80 B)
#define NTP 72               // padded B row stride in elements (144 B)
#define NSTAGE (DM / KT)     // 32 K-stages
#define ASZ (MT * KTP)       // A buffer size in elements (10240 B)
#define BSZ (KT * NTP)       // B buffer size in elements (4608 B)

// Chunked scan config
#define NC  64               // chunks along time
#define CH  (TT / NC)        // 64 steps per chunk
#define BD  (BB * DM)        // 4096 independent scalar recurrences

typedef __attribute__((ext_vector_type(16))) __bf16 v16bf;
typedef __attribute__((ext_vector_type(8)))  float  v8f;

union FragBF { v16bf v; uint4 u[2]; };

__device__ __forceinline__ unsigned short f32_to_bf16(float f) {
  union { float f; unsigned int u; } c; c.f = f;
  unsigned int lsb = (c.u >> 16) & 1u;
  c.u += 0x7fffu + lsb;                    // round-to-nearest-even
  return (unsigned short)(c.u >> 16);
}

__device__ __forceinline__ float sigmoidf_(float z) {
  return 1.0f / (1.0f + __expf(-z));
}

// Async global -> LDS 16B copy (CDNA5, ASYNCcnt-tracked).
__device__ __forceinline__ void async_cp16(unsigned lds_off, const void* gptr) {
  asm volatile("global_load_async_to_lds_b128 %0, %1, off"
               :: "v"(lds_off), "v"(gptr) : "memory");
}

__device__ __forceinline__ void wait_async_all() {
  asm volatile("s_wait_asynccnt 0" ::: "memory");
}

// ---------------------------------------------------------------------------
// 1) RMSNorm: x (B*T, D) f32 -> xn bf16
// ---------------------------------------------------------------------------
__global__ void k_rmsnorm(const float* __restrict__ x,
                          const float* __restrict__ w,
                          unsigned short* __restrict__ xn) {
  int row = blockIdx.x;
  int tid = threadIdx.x;
  const float* xr = x + (size_t)row * DM;
  float s = 0.f;
  for (int i = tid; i < DM; i += 256) { float v = xr[i]; s += v * v; }
  __shared__ float red[256];
  red[tid] = s;
  __syncthreads();
  for (int off = 128; off > 0; off >>= 1) {
    if (tid < off) red[tid] += red[tid + off];
    __syncthreads();
  }
  float scale = rsqrtf(red[0] * (1.0f / (float)DM) + RMS_EPS);
  for (int i = tid; i < DM; i += 256)
    xn[(size_t)row * DM + i] = f32_to_bf16(xr[i] * scale * w[i]);
}

// ---------------------------------------------------------------------------
// 2) Weight convert+transpose: W[n][k] f32 (x3) -> Wt[k][n_concat] bf16
//    grid (32, 32, 3), block (32, 8)
// ---------------------------------------------------------------------------
__global__ void k_wcvt(const float* __restrict__ w0,
                       const float* __restrict__ w1,
                       const float* __restrict__ w2,
                       unsigned short* __restrict__ wt) {
  __shared__ float tile[32][33];
  int z = blockIdx.z;
  const float* W = (z == 0) ? w0 : (z == 1) ? w1 : w2;
  int ntile = blockIdx.x * 32, ktile = blockIdx.y * 32;
  int tx = threadIdx.x, ty = threadIdx.y;
  for (int i = ty; i < 32; i += 8)
    tile[i][tx] = W[(size_t)(ntile + i) * DM + ktile + tx];
  __syncthreads();
  for (int i = ty; i < 32; i += 8)
    wt[(size_t)(ktile + i) * NTOT + z * DM + ntile + tx] = f32_to_bf16(tile[tx][i]);
}

// ---------------------------------------------------------------------------
// 3) Fused projection GEMM with async-LDS double buffering.
//    Block tile: 128 (m) x 64 (n), K staged 32 at a time.
//    Single loop body (unroll disabled) keeps accumulators in place; LDS
//    fragment reads index the __shared__ array directly (ds_load path).
// ---------------------------------------------------------------------------
__global__ void __launch_bounds__(256)
k_gemm_wmma(const unsigned short* __restrict__ xn,
            const unsigned short* __restrict__ wt,
            float* __restrict__ fbuf,
            float* __restrict__ vbuf,
            float* __restrict__ obuf) {
  __shared__ unsigned short smA[2 * ASZ];   // 2 x 10240 B (padded rows)
  __shared__ unsigned short smB[2 * BSZ];   // 2 x  4608 B (padded rows)

  int tid  = threadIdx.x;
  int wv   = tid >> 5;
  int lane = tid & 31;
  int lrow = lane & 15;
  int half = lane >> 4;

  int bm = blockIdx.x & 127;      // 128 m-blocks
  int bn = blockIdx.x >> 7;       // 48 n-blocks
  int m0 = bm * MT;
  int n0 = bn * NT;

  // Copy-assignment geometry (bytes): A tile is 128 rows x 64 B,
  // B tile is 32 rows x 128 B. Thread moves A:2x16B + B:1x16B per stage.
  int offs = tid * 16;
  int rA0 = offs >> 6,            cA0 = offs & 63;
  int rA1 = (offs + 4096) >> 6,   cA1 = (offs + 4096) & 63;
  int rB  = offs >> 7,            cB  = offs & 127;

  const char* gxn = (const char*)xn;
  const char* gwt = (const char*)wt;
  size_t gA0row = ((size_t)(m0 + rA0) * DM) * 2 + cA0;
  size_t gA1row = ((size_t)(m0 + rA1) * DM) * 2 + cA1;
  size_t gBrow  = ((size_t)rB * NTOT + n0) * 2 + cB;

  unsigned ldsA = (unsigned)(uintptr_t)&smA[0];
  unsigned ldsB = (unsigned)(uintptr_t)&smB[0];
  unsigned dA0 = rA0 * (KTP * 2) + cA0;
  unsigned dA1 = rA1 * (KTP * 2) + cA1;
  unsigned dB  = rB  * (NTP * 2) + cB;

  // Per-lane fragment element offsets (buffer 0).
  int afrag = (wv * 16 + lrow) * KTP + half * 8;
  int bfrag = half * 8 * NTP + lrow;

  // Prologue: stage 0 copies into buffer 0.
  async_cp16(ldsA + dA0, gxn + gA0row);
  async_cp16(ldsA + dA1, gxn + gA1row);
  async_cp16(ldsB + dB,  gwt + gBrow);

  v8f c0 = {}, c1 = {}, c2 = {}, c3 = {};

#pragma clang loop unroll(disable)
  for (int s = 0; s < NSTAGE; ++s) {
    int buf = s & 1;
    wait_async_all();        // my copies for buf are done
    __syncthreads();         // everyone's copies done; prior reads retired

    if (s + 1 < NSTAGE) {    // prefetch next stage into the other buffer
      size_t kb = (size_t)(s + 1) * KT * 2;            // K offset in bytes
      unsigned nb = (buf ^ 1);
      async_cp16(ldsA + nb * (ASZ * 2) + dA0, gxn + gA0row + kb);
      async_cp16(ldsA + nb * (ASZ * 2) + dA1, gxn + gA1row + kb);
      async_cp16(ldsB + nb * (BSZ * 2) + dB,  gwt + gBrow + kb * (NTOT / 2));
    }

    // A fragment (ISA 16-bit A 16x32 layout): lane -> row wv*16+lrow,
    // lanes<16 take K {0..7,16..23}, lanes>=16 take K {8..15,24..31}.
    int ao = afrag + buf * ASZ;
    FragBF a;
    a.u[0] = *(const uint4*)&smA[ao];
    a.u[1] = *(const uint4*)&smA[ao + 16];

    // B fragments: lane -> column n = lrow; K order same as A (8h+e, 16+8h+e-8).
    int bo = bfrag + buf * BSZ;
    FragBF b0, b1, b2, b3;
#pragma unroll
    for (int e = 0; e < 16; ++e) {
      int ko = bo + (e < 8 ? e : e + 8) * NTP;
      b0.v[e] = __builtin_bit_cast(__bf16, smB[ko]);
      b1.v[e] = __builtin_bit_cast(__bf16, smB[ko + 16]);
      b2.v[e] = __builtin_bit_cast(__bf16, smB[ko + 32]);
      b3.v[e] = __builtin_bit_cast(__bf16, smB[ko + 48]);
    }

    c0 = __builtin_amdgcn_wmma_f32_16x16x32_bf16(false, a.v, false, b0.v,
                                                 (short)0, c0, false, false);
    c1 = __builtin_amdgcn_wmma_f32_16x16x32_bf16(false, a.v, false, b1.v,
                                                 (short)0, c1, false, false);
    c2 = __builtin_amdgcn_wmma_f32_16x16x32_bf16(false, a.v, false, b2.v,
                                                 (short)0, c2, false, false);
    c3 = __builtin_amdgcn_wmma_f32_16x16x32_bf16(false, a.v, false, b3.v,
                                                 (short)0, c3, false, false);
  }

  // Epilogue: C/D layout — VGPR r: lanes<16 -> (M=r, N=lrow), lanes>=16 -> (M=r+8).
  int mbase = m0 + wv * 16 + half * 8;
  v8f accs[4] = {c0, c1, c2, c3};
#pragma unroll
  for (int j = 0; j < 4; ++j) {
    int n = n0 + j * 16 + lrow;
#pragma unroll
    for (int r = 0; r < 8; ++r) {
      int m = mbase + r;
      float val = accs[j][r];
      if (n < DM) {
        fbuf[(size_t)m * DM + n] = fmaxf(sigmoidf_(val), FORGET_LB);
      } else if (n < 2 * DM) {
        vbuf[(size_t)m * DM + (n - DM)] = val;
      } else {
        obuf[(size_t)m * DM + (n - 2 * DM)] = sigmoidf_(val);
      }
    }
  }
}

// ---------------------------------------------------------------------------
// 4) sv[m, h] = sum_e v[m, h*128 + e]  (one wave per (m,h), deterministic)
// ---------------------------------------------------------------------------
__global__ void k_svred(const float* __restrict__ vbuf, float* __restrict__ sv) {
  int gid = blockIdx.x * 256 + threadIdx.x;
  int wid = gid >> 5;                  // m*8 + h, 131072 waves
  int lane = threadIdx.x & 31;
  int m = wid >> 3, h = wid & 7;
  const float* vr = vbuf + (size_t)m * DM + h * HD;
  float s = vr[lane] + vr[lane + 32] + vr[lane + 64] + vr[lane + 96];
#pragma unroll
  for (int off = 16; off > 0; off >>= 1) s += __shfl_xor(s, off, 32);
  if (lane == 0) sv[m * NH + h] = s;
}

// ---------------------------------------------------------------------------
// 5) Chunked scan over the collapsed scalar recurrence
//    g[bd, t] = f[bd,t] * g[bd,t-1] + (1-f[bd,t]) * sv[b,h,t]
// ---------------------------------------------------------------------------
__global__ void k_scan1(const float* __restrict__ fbuf,
                        const float* __restrict__ sv,
                        float* __restrict__ abuf,
                        float* __restrict__ bbuf) {
  int bd = blockIdx.x * 256 + threadIdx.x;   // 0..4095
  int c = blockIdx.y;
  int b = bd >> 10, d = bd & 1023, h = d >> 7;
  float A = 1.f, g = 0.f;
  int mbase = b * TT + c * CH;
  for (int j = 0; j < CH; ++j) {
    int m = mbase + j;
    float fv = fbuf[(size_t)m * DM + d];
    float s = sv[m * NH + h];
    g = fv * g + (1.f - fv) * s;
    A *= fv;
  }
  abuf[c * BD + bd] = A;
  bbuf[c * BD + bd] = g;
}

__global__ void k_scan2(const float* __restrict__ abuf,
                        const float* __restrict__ bbuf,
                        float* __restrict__ gstart) {
  int bd = blockIdx.x * 256 + threadIdx.x;
  float g = 0.f;
  for (int c = 0; c < NC; ++c) {
    gstart[c * BD + bd] = g;
    g = abuf[c * BD + bd] * g + bbuf[c * BD + bd];
  }
}

__global__ void k_scan3(const float* __restrict__ fbuf,
                        const float* __restrict__ obuf,
                        const float* __restrict__ sv,
                        const float* __restrict__ gstart,
                        const float* __restrict__ x,
                        float* __restrict__ out) {
  int bd = blockIdx.x * 256 + threadIdx.x;
  int c = blockIdx.y;
  int b = bd >> 10, d = bd & 1023, h = d >> 7;
  float g = gstart[c * BD + bd];
  int mbase = b * TT + c * CH;
  for (int j = 0; j < CH; ++j) {
    int m = mbase + j;
    size_t idx = (size_t)m * DM + d;
    float fv = fbuf[idx];
    float s = sv[m * NH + h];
    g = fv * g + (1.f - fv) * s;
    out[idx] = x[idx] + obuf[idx] * g;
  }
}

// ---------------------------------------------------------------------------
extern "C" void kernel_launch(void* const* d_in, const int* in_sizes, int n_in,
                              void* d_out, int out_size, void* d_ws, size_t ws_size,
                              hipStream_t stream) {
  (void)in_sizes; (void)n_in; (void)out_size; (void)ws_size;
  const float* x  = (const float*)d_in[0];
  const float* nw = (const float*)d_in[1];
  const float* wf = (const float*)d_in[2];
  const float* wi = (const float*)d_in[3];
  const float* wo = (const float*)d_in[4];
  float* out = (float*)d_out;

  char* ws = (char*)d_ws;
  size_t off = 0;
  unsigned short* xn = (unsigned short*)(ws + off); off += (size_t)MROWS * DM * 2;  // 32 MB
  unsigned short* wt = (unsigned short*)(ws + off); off += (size_t)DM * NTOT * 2;   //  6 MB
  float* fbuf = (float*)(ws + off); off += (size_t)MROWS * DM * 4;                  // 64 MB
  float* vbuf = (float*)(ws + off); off += (size_t)MROWS * DM * 4;                  // 64 MB
  float* obuf = (float*)(ws + off); off += (size_t)MROWS * DM * 4;                  // 64 MB
  float* sv   = (float*)(ws + off); off += (size_t)MROWS * NH * 4;                  // 0.5 MB
  // Scan scratch aliases vbuf (vbuf is dead after k_svred; same-stream ordering).
  float* abuf   = vbuf;
  float* bbuf   = vbuf + (size_t)NC * BD;
  float* gstart = vbuf + (size_t)2 * NC * BD;

  k_rmsnorm<<<MROWS, 256, 0, stream>>>(x, nw, xn);
  k_wcvt<<<dim3(32, 32, 3), dim3(32, 8), 0, stream>>>(wf, wi, wo, wt);
  k_gemm_wmma<<<(MROWS / MT) * (NTOT / NT), 256, 0, stream>>>(xn, wt, fbuf, vbuf, obuf);
  k_svred<<<(MROWS * NH) / 8, 256, 0, stream>>>(vbuf, sv);
  k_scan1<<<dim3(BD / 256, NC), 256, 0, stream>>>(fbuf, sv, abuf, bbuf);
  k_scan2<<<BD / 256, 256, 0, stream>>>(abuf, bbuf, gstart);
  k_scan3<<<dim3(BD / 256, NC), 256, 0, stream>>>(fbuf, obuf, sv, gstart, x, out);
}
